// Mamba2Chunked_52029233824001
// MI455X (gfx1250) — compile-verified, matrix operand order checked by reading, not run on hardware
//
#include <hip/hip_runtime.h>
#include <hip/hip_bf16.h>
#include <math.h>
#include <stdint.h>

// ---------------- problem dims ----------------
#define D_MODEL   1024
#define D_STATE   64
#define D_CONV    4
#define HEADDIM   128
#define NHEADS    16
#define CHUNKLEN  256
#define D_INNER   2048            // NHEADS*HEADDIM
#define D_IN_PROJ 4240            // 2*D_INNER + 2*D_STATE + NHEADS
#define CONV_DIM  2176            // D_INNER + 2*D_STATE
#define BATCH     4
#define SEQ       2048
#define NCHUNKS   8
#define ROWS      (BATCH*SEQ)     // 8192
#define CORRW     (D_INNER + NHEADS + 2*D_STATE)  // 2192

// ---------------- types ----------------
typedef __attribute__((ext_vector_type(16))) __bf16          v16bf;
typedef __attribute__((ext_vector_type(8)))  float           v8f;
typedef __attribute__((ext_vector_type(8)))  unsigned short  us8;
typedef __attribute__((ext_vector_type(4)))  unsigned int    u32x4;
typedef __attribute__((ext_vector_type(8)))  int             i32x8;
typedef __attribute__((ext_vector_type(4)))  int             i32x4;
struct us16 { us8 lo, hi; };   // 32B, bit-identical to v16bf

#if __has_builtin(__builtin_amdgcn_tensor_load_to_lds) && __has_builtin(__builtin_amdgcn_s_wait_tensorcnt)
#define HAVE_TDM 1
#else
#define HAVE_TDM 0
#endif

__device__ __forceinline__ unsigned short f2bf_u(float f) {
  unsigned u = __builtin_bit_cast(unsigned, f);
  u += 0x7FFFu + ((u >> 16) & 1u);          // round-to-nearest-even
  return (unsigned short)(u >> 16);
}
__device__ __forceinline__ float sigmoidf_(float x) { return 1.0f / (1.0f + __expf(-x)); }
__device__ __forceinline__ float siluf_(float x)    { return x * sigmoidf_(x); }
__device__ __forceinline__ float softplusf_(float x){ return (x > 20.0f) ? x : log1pf(__expf(x)); }

#if HAVE_TDM
// ---------------------------------------------------------------------
// Issue one TDM 2D tile load: bf16 tile (tile_k x tile_rows) from a
// row-major [rows][stride] bf16 tensor into LDS (packed [rows][tile_k]).
// D# built per CDNA5 ISA 8.3/8.4: group0 = {flags, lds_addr, global_addr,
// type=2}; group1 carries data_size=2B, tensor dims, tile dims, stride.
// Rows beyond rows_remaining are zero-filled by TDM OOB handling.
// This toolchain exposes the 6-arg builtin:
//   (u32x4 g0, i32x8 g1, i32x4 g2, i32x4 g3, i32x8, i32 cpol)
// ---------------------------------------------------------------------
__device__ __forceinline__ void tdm_load_tile_bf16(
    const unsigned short* gtile, unsigned ldsoff,
    unsigned k_remaining, unsigned rows_remaining,
    unsigned stride_elems, unsigned tile_k, unsigned tile_rows) {
  unsigned long long ga = (unsigned long long)(uintptr_t)gtile;
  u32x4 g0;
  g0[0] = 1u;                                            // count=1, user desc
  g0[1] = ldsoff;                                        // lds_addr (bytes)
  g0[2] = (unsigned)ga;                                  // global_addr[31:0]
  g0[3] = (unsigned)((ga >> 32) & 0x01FFFFFFu) | (2u << 30); // addr[56:32], type=2
  i32x8 g1;
  g1[0] = (int)(1u << 16);                               // wg_mask=0, data_size=1 (2B)
  g1[1] = (int)((k_remaining & 0xFFFFu) << 16);          // tensor_dim0[15:0]
  g1[2] = (int)(((k_remaining >> 16) & 0xFFFFu) |
                ((rows_remaining & 0xFFFFu) << 16));     // dim0[31:16] | dim1[15:0]
  g1[3] = (int)(((rows_remaining >> 16) & 0xFFFFu) |
                ((tile_k & 0xFFFFu) << 16));             // dim1[31:16] | tile_dim0
  g1[4] = (int)(tile_rows & 0xFFFFu);                    // tile_dim1 (tile_dim2=0)
  g1[5] = (int)stride_elems;                             // tensor_dim0_stride[31:0]
  g1[6] = 0;                                             // stride[47:32] | dim1_stride lo
  g1[7] = 0;
  i32x4 z4 = {0, 0, 0, 0};
  i32x8 z8 = {0, 0, 0, 0, 0, 0, 0, 0};
  __builtin_amdgcn_tensor_load_to_lds(g0, g1, z4, z4, z8, 0);
}
#endif

__device__ __forceinline__ unsigned lds_off_of(const void* p) {
  return (unsigned)(unsigned long long)(uintptr_t)p;     // LDS flat addr low 32 bits
}

// =====================================================================
// bf16 WMMA GEMM: C[M,N] = A[M,K] * Bt[N,K]^T, A/Bt bf16, C f32.
// Block = 256 threads (8 waves), tile 128x64, wave tile 32x32, K step 32.
// Tiles staged to LDS by the Tensor Data Mover (double-buffered, one
// issuing wave, s_wait_tensorcnt + workgroup barrier), overlapping WMMA.
// =====================================================================
#define BM 128
#define BN 64
#define BK 32

__global__ __launch_bounds__(256)
void gemm_bf16wmma(const unsigned short* __restrict__ A,
                   const unsigned short* __restrict__ Bt,
                   float* __restrict__ C, int M, int N, int K) {
  __shared__ __align__(16) unsigned short sA[2][BM * BK];
  __shared__ __align__(16) unsigned short sB[2][BN * BK];

  const int tid  = threadIdx.x;
  const int lane = tid & 31;
  const int wid  = tid >> 5;
  const int wm   = (wid & 3) * 32;
  const int wn   = (wid >> 2) * 32;
  const int m0   = blockIdx.y * BM;
  const int n0   = blockIdx.x * BN;
  const int frow = lane & 15;
  const int kb   = (lane >> 4) * 8;

  v8f acc[2][2];
  #pragma unroll
  for (int i = 0; i < 2; ++i)
    #pragma unroll
    for (int j = 0; j < 2; ++j)
      acc[i][j] = (v8f){0.f,0.f,0.f,0.f,0.f,0.f,0.f,0.f};

  const int ktiles = K / BK;

#if HAVE_TDM
  if (tid < 32) {   // one wave drives the DMA engine
    tdm_load_tile_bf16(A  + (size_t)m0 * K, lds_off_of(&sA[0][0]),
                       (unsigned)K, (unsigned)(M - m0), (unsigned)K, BK, BM);
    tdm_load_tile_bf16(Bt + (size_t)n0 * K, lds_off_of(&sB[0][0]),
                       (unsigned)K, (unsigned)(N - n0), (unsigned)K, BK, BN);
  }
#endif

  for (int kt = 0; kt < ktiles; ++kt) {
    const int cur = kt & 1;
#if HAVE_TDM
    if (tid < 32) __builtin_amdgcn_s_wait_tensorcnt(0);  // tile kt landed
    __syncthreads();
    if (kt + 1 < ktiles && tid < 32) {                   // prefetch tile kt+1
      const int nxt = cur ^ 1;
      tdm_load_tile_bf16(A  + (size_t)m0 * K + (size_t)(kt + 1) * BK,
                         lds_off_of(&sA[nxt][0]),
                         (unsigned)(K - (kt + 1) * BK), (unsigned)(M - m0),
                         (unsigned)K, BK, BM);
      tdm_load_tile_bf16(Bt + (size_t)n0 * K + (size_t)(kt + 1) * BK,
                         lds_off_of(&sB[nxt][0]),
                         (unsigned)(K - (kt + 1) * BK), (unsigned)(N - n0),
                         (unsigned)K, BK, BN);
    }
#else
    // fallback: vector copy global->LDS (b128 both sides)
    #pragma unroll
    for (int i = 0; i < 2; ++i) {                        // A: 512 us8 chunks
      int idx = tid + i * 256;
      int r = idx >> 2, seg = idx & 3;
      *(us8*)&sA[cur][r * BK + seg * 8] =
          *(const us8*)&A[(size_t)(m0 + r) * K + (size_t)kt * BK + seg * 8];
    }
    {                                                     // B: 256 us8 chunks
      int r = tid >> 2, seg = tid & 3;
      us8 v = (us8){0,0,0,0,0,0,0,0};
      if (n0 + r < N)
        v = *(const us8*)&Bt[(size_t)(n0 + r) * K + (size_t)kt * BK + seg * 8];
      *(us8*)&sB[cur][r * BK + seg * 8] = v;
    }
    __syncthreads();
#endif

    // ---- fragments from LDS (two ds_load_b128 each) ----
    v16bf afrag[2], bfrag[2];
    #pragma unroll
    for (int fm = 0; fm < 2; ++fm) {
      const unsigned short* p = &sA[cur][(wm + fm * 16 + frow) * BK + kb];
      us16 t; t.lo = *(const us8*)p; t.hi = *(const us8*)(p + 16);
      afrag[fm] = __builtin_bit_cast(v16bf, t);
    }
    #pragma unroll
    for (int fn = 0; fn < 2; ++fn) {
      const unsigned short* p = &sB[cur][(wn + fn * 16 + frow) * BK + kb];
      us16 t; t.lo = *(const us8*)p; t.hi = *(const us8*)(p + 16);
      bfrag[fn] = __builtin_bit_cast(v16bf, t);
    }

    #pragma unroll
    for (int fm = 0; fm < 2; ++fm)
      #pragma unroll
      for (int fn = 0; fn < 2; ++fn)
        acc[fm][fn] = __builtin_amdgcn_wmma_f32_16x16x32_bf16(
            false, afrag[fm], false, bfrag[fn], (short)0, acc[fm][fn], false, false);

    __syncthreads();   // tile cur free for reuse at kt+2
  }

  const int rhalf = (lane >> 4) * 8;
  const int ccol  = lane & 15;
  #pragma unroll
  for (int fm = 0; fm < 2; ++fm)
    #pragma unroll
    for (int fn = 0; fn < 2; ++fn)
      #pragma unroll
      for (int r = 0; r < 8; ++r) {
        int gr = m0 + wm + fm * 16 + r + rhalf;
        int gc = n0 + wn + fn * 16 + ccol;
        if (gc < N) C[(size_t)gr * N + gc] = acc[fm][fn][r];
      }
}

// =====================================================================
// f32 -> bf16 converters (packed dword stores)
// =====================================================================
__global__ __launch_bounds__(256)
void cvt_bf16(const float* __restrict__ in, unsigned* __restrict__ out, size_t npairs) {
  size_t i = (size_t)blockIdx.x * 256 + threadIdx.x;
  if (i >= npairs) return;
  unsigned lo = f2bf_u(in[2 * i]);
  unsigned hi = f2bf_u(in[2 * i + 1]);
  out[i] = lo | (hi << 16);
}

// out[n][k] = bf16(in[k][n]) : row-major [K][N] f32 -> [N][K] bf16
__global__ __launch_bounds__(256)
void cvt_bf16_t(const float* __restrict__ in, unsigned* __restrict__ out, int N, int K) {
  size_t i = (size_t)blockIdx.x * 256 + threadIdx.x;
  const int khalf = K >> 1;
  if (i >= (size_t)N * khalf) return;
  int n  = (int)(i / khalf);
  int k2 = (int)(i % khalf) * 2;
  unsigned lo = f2bf_u(in[(size_t)k2 * N + n]);
  unsigned hi = f2bf_u(in[(size_t)(k2 + 1) * N + n]);
  out[i] = lo | (hi << 16);
}

// =====================================================================
// Depthwise causal conv (width 4) + SiLU over the xBC slice of zxbcdt.
// =====================================================================
__global__ __launch_bounds__(256)
void conv_silu(const float* __restrict__ zx, float* __restrict__ xbc,
               const float* __restrict__ cw, const float* __restrict__ cb) {
  size_t idx = (size_t)blockIdx.x * 256 + threadIdx.x;
  const size_t total = (size_t)ROWS * CONV_DIM;
  if (idx >= total) return;
  int c     = (int)(idx % CONV_DIM);
  size_t rl = idx / CONV_DIM;
  int l = (int)(rl % SEQ);
  int b = (int)(rl / SEQ);
  float acc = cb[c];
  #pragma unroll
  for (int i = 0; i < D_CONV; ++i) {
    int li = l - (D_CONV - 1) + i;
    float xv = (li >= 0) ? zx[((size_t)(b * SEQ + li)) * D_IN_PROJ + D_INNER + c] : 0.0f;
    acc += xv * cw[c * D_CONV + i];
  }
  xbc[idx] = siluf_(acc);
}

// =====================================================================
// Chunk-boundary low-rank corrections: corr[b] = s_prev[b] @ [Rx|Rdt|RB|RC]
// =====================================================================
__global__ __launch_bounds__(256)
void chunk_corr(const float* __restrict__ s_prev, const float* __restrict__ Rx,
                const float* __restrict__ Rdt, const float* __restrict__ RB,
                const float* __restrict__ RC, float* __restrict__ corr) {
  __shared__ float sp[D_INNER];
  const int b = blockIdx.y;
  const int j = blockIdx.x * 256 + threadIdx.x;
  for (int i = threadIdx.x; i < D_INNER; i += 256) sp[i] = s_prev[b * D_INNER + i];
  __syncthreads();
  if (j >= CORRW) return;
  const float* Rm; int w, col;
  if (j < D_INNER)                         { Rm = Rx;  w = D_INNER; col = j; }
  else if (j < D_INNER + NHEADS)           { Rm = Rdt; w = NHEADS;  col = j - D_INNER; }
  else if (j < D_INNER + NHEADS + D_STATE) { Rm = RB;  w = D_STATE; col = j - D_INNER - NHEADS; }
  else                                     { Rm = RC;  w = D_STATE; col = j - D_INNER - NHEADS - D_STATE; }
  float acc = 0.f;
  for (int i = 0; i < D_INNER; ++i) acc += sp[i] * Rm[(size_t)i * w + col];
  corr[b * CORRW + j] = acc;
}

// =====================================================================
// Sequential scan over one chunk; block=(b,h), lane=p; 64-wide state in VGPRs
// =====================================================================
__global__ __launch_bounds__(128)
void chunk_scan(const float* __restrict__ zx, const float* __restrict__ xbc,
                const float* __restrict__ corr, const float* __restrict__ dt_bias,
                const float* __restrict__ A_log, const float* __restrict__ Dv,
                float* __restrict__ ys, float* __restrict__ cstate,
                float* __restrict__ s_prev, int kchunk) {
  const int h = blockIdx.x;
  const int b = blockIdx.y;
  const int p = threadIdx.x;
  __shared__ float sB[D_STATE];
  __shared__ float sC[D_STATE];

  const float* cb = corr + b * CORRW;
  const float Ah  = -__expf(A_log[h]);
  const float Dh  = Dv[h];
  const float dtb = dt_bias[h];
  const float dtc = cb[D_INNER + h];
  const float xc  = cb[h * HEADDIM + p];

  float cst[D_STATE];
  const size_t coff = (((size_t)(b * NHEADS + h)) * HEADDIM + p) * D_STATE;
  #pragma unroll
  for (int n = 0; n < D_STATE; ++n) cst[n] = cstate[coff + n];

  float snew = 0.f;
  for (int t = 0; t < CHUNKLEN; ++t) {
    const int l = kchunk * CHUNKLEN + t;
    const size_t rb = (size_t)(b * SEQ + l);
    if (p < D_STATE)
      sB[p] = xbc[rb * CONV_DIM + D_INNER + p] + cb[D_INNER + NHEADS + p];
    else if (p < 2 * D_STATE)
      sC[p - D_STATE] = xbc[rb * CONV_DIM + D_INNER + D_STATE + (p - D_STATE)]
                        + cb[D_INNER + NHEADS + D_STATE + (p - D_STATE)];
    __syncthreads();

    float dtr   = zx[rb * D_IN_PROJ + (D_IN_PROJ - NHEADS) + h] + dtc + dtb;
    float dtv   = softplusf_(dtr);
    float decay = __expf(Ah * dtv);
    float x     = xbc[rb * CONV_DIM + h * HEADDIM + p] + xc;
    float dx    = dtv * x;

    float y = 0.f;
    #pragma unroll
    for (int n = 0; n < D_STATE; ++n) {
      cst[n] = decay * cst[n] + dx * sB[n];
      y += cst[n] * sC[n];
    }
    y += Dh * x;
    ys[rb * D_INNER + h * HEADDIM + p] = y;
    snew += y;
    __syncthreads();
  }

  #pragma unroll
  for (int n = 0; n < D_STATE; ++n) cstate[coff + n] = cst[n];
  s_prev[b * D_INNER + h * HEADDIM + p] = snew;
}

// =====================================================================
// y * silu(z), RMSNorm, * norm_w -> packed bf16 (feeds GEMM2 directly)
// =====================================================================
__global__ __launch_bounds__(256)
void gate_rmsnorm(const float* __restrict__ ys, const float* __restrict__ zx,
                  const float* __restrict__ norm_w, unsigned short* __restrict__ ynbf) {
  __shared__ float red[256];
  const size_t row = blockIdx.x;
  const int tid = threadIdx.x;
  const int d0 = tid * 8;
  float g[8];
  float ss = 0.f;
  #pragma unroll
  for (int i = 0; i < 8; ++i) {
    int d = d0 + i;
    float y = ys[row * D_INNER + d];
    float z = zx[row * D_IN_PROJ + d];
    float v = y * siluf_(z);
    g[i] = v;
    ss += v * v;
  }
  red[tid] = ss;
  __syncthreads();
  for (int s = 128; s > 0; s >>= 1) {
    if (tid < s) red[tid] += red[tid + s];
    __syncthreads();
  }
  float scale = rsqrtf(red[0] / (float)D_INNER + 1e-5f);
  unsigned* o32 = (unsigned*)(ynbf + row * D_INNER);
  #pragma unroll
  for (int j = 0; j < 4; ++j) {
    float a = g[2 * j]     * scale * norm_w[d0 + 2 * j];
    float c = g[2 * j + 1] * scale * norm_w[d0 + 2 * j + 1];
    o32[(d0 >> 1) + j] = (unsigned)f2bf_u(a) | ((unsigned)f2bf_u(c) << 16);
  }
}

__global__ void zero_ws(float* __restrict__ p, size_t n) {
  size_t i = (size_t)blockIdx.x * 256 + threadIdx.x;
  if (i < n) p[i] = 0.f;
}

// =====================================================================
// Host-side orchestration
// =====================================================================
extern "C" void kernel_launch(void* const* d_in, const int* in_sizes, int n_in,
                              void* d_out, int out_size, void* d_ws, size_t ws_size,
                              hipStream_t stream) {
  const float* u       = (const float*)d_in[0];
  const float* W_in    = (const float*)d_in[1];
  const float* conv_w  = (const float*)d_in[2];
  const float* conv_b  = (const float*)d_in[3];
  const float* dt_bias = (const float*)d_in[4];
  const float* A_log   = (const float*)d_in[5];
  const float* Dv      = (const float*)d_in[6];
  const float* Rx      = (const float*)d_in[7];
  const float* Rdt     = (const float*)d_in[8];
  const float* RB      = (const float*)d_in[9];
  const float* RC      = (const float*)d_in[10];
  const float* norm_w  = (const float*)d_in[11];
  const float* W_out   = (const float*)d_in[12];
  float* out = (float*)d_out;

  float* ws = (float*)d_ws;
  const size_t o_zx     = 0;
  const size_t o_xbc    = o_zx     + (size_t)ROWS * D_IN_PROJ;
  const size_t o_ys     = o_xbc    + (size_t)ROWS * CONV_DIM;
  const size_t o_sprev  = o_ys     + (size_t)ROWS * D_INNER;
  const size_t o_corr   = o_sprev  + (size_t)BATCH * D_INNER;
  const size_t o_cstate = o_corr   + (size_t)BATCH * CORRW;
  const size_t o_ubf    = o_cstate + (size_t)BATCH * NHEADS * HEADDIM * D_STATE;
  const size_t o_wint   = o_ubf    + ((size_t)ROWS * D_MODEL) / 2;      // bf16 halves
  const size_t o_woutt  = o_wint   + ((size_t)D_IN_PROJ * D_MODEL) / 2;
  const size_t total    = o_woutt  + ((size_t)D_MODEL * D_INNER) / 2;
  if (ws_size < total * sizeof(float)) return;

  float* zx     = ws + o_zx;
  float* xbc    = ws + o_xbc;
  float* ysbuf  = ws + o_ys;
  float* sprev  = ws + o_sprev;
  float* corr   = ws + o_corr;
  float* cstate = ws + o_cstate;
  unsigned short* ubf   = (unsigned short*)(ws + o_ubf);
  unsigned short* wint  = (unsigned short*)(ws + o_wint);
  unsigned short* woutt = (unsigned short*)(ws + o_woutt);
  unsigned short* ynbf  = (unsigned short*)xbc;   // xbc dead after scan; reuse

  // zero s_prev + corr + cstate (contiguous)
  const size_t nz = o_ubf - o_sprev;
  zero_ws<<<dim3((unsigned)((nz + 255) / 256)), 256, 0, stream>>>(sprev, nz);

  // 0) precision staging: f32 -> bf16 (u row-major; weights transposed [N][K])
  const size_t upairs = (size_t)ROWS * D_MODEL / 2;
  cvt_bf16<<<dim3((unsigned)((upairs + 255) / 256)), 256, 0, stream>>>(u, (unsigned*)ubf, upairs);
  {
    size_t n = (size_t)D_IN_PROJ * (D_MODEL / 2);
    cvt_bf16_t<<<dim3((unsigned)((n + 255) / 256)), 256, 0, stream>>>(W_in, (unsigned*)wint, D_IN_PROJ, D_MODEL);
  }
  {
    size_t n = (size_t)D_MODEL * (D_INNER / 2);
    cvt_bf16_t<<<dim3((unsigned)((n + 255) / 256)), 256, 0, stream>>>(W_out, (unsigned*)woutt, D_MODEL, D_INNER);
  }

  // 1) zxbcdt = u @ W_in  (8192x1024 @ 1024x4240), TDM-staged bf16 WMMA
  gemm_bf16wmma<<<dim3((D_IN_PROJ + BN - 1) / BN, ROWS / BM), 256, 0, stream>>>(
      ubf, wint, zx, ROWS, D_IN_PROJ, D_MODEL);

  // 2) causal depthwise conv + SiLU
  const size_t nconv = (size_t)ROWS * CONV_DIM;
  conv_silu<<<dim3((unsigned)((nconv + 255) / 256)), 256, 0, stream>>>(zx, xbc, conv_w, conv_b);

  // 3) chunked scan
  for (int k = 0; k < NCHUNKS; ++k) {
    chunk_corr<<<dim3((CORRW + 255) / 256, BATCH), 256, 0, stream>>>(
        sprev, Rx, Rdt, RB, RC, corr);
    chunk_scan<<<dim3(NHEADS, BATCH), 128, 0, stream>>>(
        zx, xbc, corr, dt_bias, A_log, Dv, ysbuf, cstate, sprev, k);
  }

  // 4) gate + RMSNorm -> bf16
  gate_rmsnorm<<<dim3(ROWS), 256, 0, stream>>>(ysbuf, zx, norm_w, ynbf);

  // 5) out = ynorm @ W_out  (8192x2048 @ 2048x1024)
  gemm_bf16wmma<<<dim3(D_MODEL / BN, ROWS / BM), 256, 0, stream>>>(
      ynbf, woutt, out, ROWS, D_MODEL, D_INNER);
}
